// CrossAttenHead_41068477284910
// MI455X (gfx1250) — matrix operand order repaired and checked by reading, not compiled, not run on hardware
//
#include <hip/hip_runtime.h>
#include <hip/hip_bf16.h>
#include <math.h>

// ---------------------------------------------------------------------------
// CrossAttenHead on MI455X (gfx1250): all matmuls via v_wmma_f32_16x16x32_f16.
// f16 staging of features/mask keeps the big-GEMM working set L2-resident.
// ---------------------------------------------------------------------------

typedef __attribute__((ext_vector_type(16))) _Float16 v16h;
typedef __attribute__((ext_vector_type(8)))  _Float16 v8h;
typedef __attribute__((ext_vector_type(8)))  float    v8f;

#define BB   8
#define CHN  256
#define HW   16384
#define NP   100
#define NPAD 112
#define NPK  103
#define FFN  2048
#define NCLS 151

// ---------------------------------------------------------------------------
// Generic batched WMMA linear:  out[b] = X[b] (Mpad x K, f16) * WT[b]^T + bias
// WT is stored (Npad x K) so each lane's B-fragment halves are contiguous in K.
// One 16x16 output tile per wave, 8 waves per block. K multiple of 32.
// A-frag (16x32 f16, lane = (g,m)): halves[h]   =X[m, k+16g+h]
// B-frag (32x16 f16, lane = (g,n)): halves[h]   =WT[n, k+16g+h]
// D (16x16 f32, lane = (g,n)):      acc[v] = out[v+8g, n]
// ---------------------------------------------------------------------------
__global__ __launch_bounds__(256) void wmma_lin_k(
    const _Float16* __restrict__ X, const _Float16* __restrict__ WT,
    const float* __restrict__ bias, float* __restrict__ out,
    _Float16* __restrict__ out16,
    int Mtiles, int Ntiles, int Mout, int K, int Nout,
    long sxb, long swb, long sob, int relu)
{
  int wave = threadIdx.x >> 5;
  int lane = threadIdx.x & 31;
  int tiles = Mtiles * Ntiles;
  int tile = blockIdx.x * 8 + wave;
  if (tile >= tiles) return;                 // uniform per wave
  int mt = tile / Ntiles;
  int nt = tile - mt * Ntiles;
  long b = blockIdx.z;
  const _Float16* arow = X  + b * sxb + (long)(mt * 16 + (lane & 15)) * K;
  const _Float16* brow = WT + b * swb + (long)(nt * 16 + (lane & 15)) * K;
  int g = lane >> 4;
  v8f acc = {0.f, 0.f, 0.f, 0.f, 0.f, 0.f, 0.f, 0.f};
  for (int k = 0; k < K; k += 32) {
    v16h a = *(const v16h*)(arow + k + 16 * g);
    v16h bm = *(const v16h*)(brow + k + 16 * g);
    acc = __builtin_amdgcn_wmma_f32_16x16x32_f16(false, a, false, bm,
                                                 (short)0, acc, false, false);
  }
  int col = nt * 16 + (lane & 15);
  if (col >= Nout) return;                   // per-lane, after all WMMAs
  float bv = bias ? bias[col] : 0.f;
  int r0 = mt * 16 + g * 8;
#pragma unroll
  for (int v = 0; v < 8; ++v) {
    int r = r0 + v;
    if (r < Mout) {
      float val = acc[v] + bv;
      if (relu) val = fmaxf(val, 0.f);
      if (out)   out  [b * sob + (long)r * Nout + col] = val;
      if (out16) out16[b * sob + (long)r * Nout + col] = (_Float16)val;
    }
  }
}

// ---- staging kernels -------------------------------------------------------

// mask_preds (B,100,HW) f32 -> hard16 (B,112,HW) f16 (sigmoid(x)>0.5 <=> x>0)
__global__ void binarize_k(const float* __restrict__ mp, _Float16* __restrict__ hard)
{
  long i = (long)blockIdx.x * 256 + threadIdx.x;
  if (i >= (long)BB * NPAD * HW) return;
  long hw = i % HW; long r = (i / HW) % NPAD; long b = i / ((long)HW * NPAD);
  _Float16 v = (_Float16)0.f;
  if (r < NP) v = (mp[((long)b * NP + r) * HW + hw] > 0.f) ? (_Float16)1.f
                                                           : (_Float16)0.f;
  hard[i] = v;
}

__global__ void cvt_k(const float* __restrict__ s, _Float16* __restrict__ d, long n)
{
  long i = (long)blockIdx.x * 256 + threadIdx.x;
  if (i < n) d[i] = (_Float16)s[i];
}

// features (B,256,HW) f32 -> featT16 (B,HW,256) f16, LDS-tiled transpose
__global__ __launch_bounds__(256) void transp_cvt_k(const float* __restrict__ src,
                                                    _Float16* __restrict__ dst)
{
  __shared__ float tile[32][33];
  int b = blockIdx.z, hw0 = blockIdx.x * 32, c0 = blockIdx.y * 32;
  int tx = threadIdx.x & 31, ty = threadIdx.x >> 5;
  const float* s = src + ((long)b * CHN + c0) * HW + hw0;
#pragma unroll
  for (int i = 0; i < 4; ++i)
    tile[ty + 8 * i][tx] = s[(long)(ty + 8 * i) * HW + tx];
  __syncthreads();
  _Float16* d = dst + ((long)b * HW + hw0) * CHN + c0;
#pragma unroll
  for (int i = 0; i < 4; ++i)
    d[(long)(ty + 8 * i) * CHN + tx] = (_Float16)tile[tx][ty + 8 * i];
}

// weight -> f16, optionally transposed, N-padded with zeros: dst (Npad x Kd)
__global__ void prepw_k(const float* __restrict__ src, _Float16* __restrict__ dst,
                        int N, int Kd, int Npad, int transp)
{
  long i = (long)blockIdx.x * 256 + threadIdx.x;
  if (i >= (long)Npad * Kd) return;
  int k = (int)(i % Kd); int n = (int)(i / Kd);
  float v = 0.f;
  if (n < N) v = transp ? src[(long)k * N + n] : src[(long)n * Kd + k];
  dst[i] = (_Float16)v;
}

// (B,100,256) f32 -> (B,112,256) f16 zero-padded rows
__global__ void padcvt_k(const float* __restrict__ src, _Float16* __restrict__ dst)
{
  long i = (long)blockIdx.x * 256 + threadIdx.x;
  if (i >= (long)BB * NPAD * CHN) return;
  int c = (int)(i % CHN); int r = (int)((i / CHN) % NPAD); int b = (int)(i / (NPAD * CHN));
  dst[i] = (r < NP) ? (_Float16)src[((long)b * NP + r) * CHN + c] : (_Float16)0.f;
}

// ---- dysep pieces ----------------------------------------------------------

// out[bm,c] = relu(dw0*v[c-1] + dw1*v[c] + dw2*v[c+1]), zero-padded channels
__global__ __launch_bounds__(256) void conv_k(const float* __restrict__ w,
                                              const float* __restrict__ val,
                                              float* __restrict__ out)
{
  int bm = blockIdx.x, c = threadIdx.x;
  const float* v = val + (long)bm * CHN;
  float d0 = w[(long)bm * NPK + 0], d1 = w[(long)bm * NPK + 1], d2 = w[(long)bm * NPK + 2];
  float l = (c > 0) ? v[c - 1] : 0.f;
  float r = (c < CHN - 1) ? v[c + 1] : 0.f;
  out[(long)bm * CHN + c] = fmaxf(d0 * l + d1 * v[c] + d2 * r, 0.f);
}

// out[b,m,c] = sum_n pw[b,m,n] * t[b,n,c]
__global__ __launch_bounds__(256) void mix_k(const float* __restrict__ w,
                                             const float* __restrict__ t,
                                             float* __restrict__ out)
{
  int bm = blockIdx.x, c = threadIdx.x;
  int b = bm / NP;
  const float* pw = w + (long)bm * NPK + 3;
  const float* tb = t + (long)b * NP * CHN;
  float s = 0.f;
  for (int n = 0; n < NP; ++n) s += pw[n] * tb[(long)n * CHN + c];
  out[(long)bm * CHN + c] = s;
}

// ---- layernorm over C=256 (fused residual / relu / f16 sideband) -----------
__global__ __launch_bounds__(256) void ln_k(const float* __restrict__ x,
                                            const float* __restrict__ res,
                                            const float* __restrict__ g,
                                            const float* __restrict__ bt,
                                            float* __restrict__ out,
                                            _Float16* __restrict__ out16, int relu)
{
  __shared__ float red[256];
  __shared__ float mv[2];
  int r = blockIdx.x, c = threadIdx.x;
  long idx = (long)r * CHN + c;
  float v = x[idx] + (res ? res[idx] : 0.f);
  red[c] = v; __syncthreads();
  for (int s = 128; s > 0; s >>= 1) { if (c < s) red[c] += red[c + s]; __syncthreads(); }
  if (c == 0) mv[0] = red[0] * (1.f / CHN);
  __syncthreads();
  float d = v - mv[0];
  red[c] = d * d; __syncthreads();
  for (int s = 128; s > 0; s >>= 1) { if (c < s) red[c] += red[c + s]; __syncthreads(); }
  if (c == 0) mv[1] = red[0] * (1.f / CHN);
  __syncthreads();
  float y = d * rsqrtf(mv[1] + 1e-5f) * g[c] + bt[c];
  if (relu) y = fmaxf(y, 0.f);
  if (out)   out[idx] = y;
  if (out16) out16[idx] = (_Float16)y;
}

// ---- attention: one block per (b,h,n), 100 keys, DH=32 ---------------------
__global__ __launch_bounds__(128) void attn_k(const float* __restrict__ qkv,
                                              _Float16* __restrict__ o16)
{
  __shared__ float q[32], lg[NP], red[128], sinv;
  int bid = blockIdx.x, t = threadIdx.x;
  int n = bid % NP, h = (bid / NP) & 7, b = bid / (NP * 8);
  long base = (long)b * NP * 768;
  if (t < 32) q[t] = qkv[base + (long)n * 768 + h * 32 + t];
  __syncthreads();
  if (t < NP) {
    const float* kr = qkv + base + (long)t * 768 + 256 + h * 32;
    float s = 0.f;
    for (int d = 0; d < 32; ++d) s += q[d] * kr[d];
    lg[t] = s * 0.17677669529663687f;           // 1/sqrt(32)
  }
  __syncthreads();
  red[t] = (t < NP) ? lg[t] : -3.4e38f; __syncthreads();
  for (int s = 64; s > 0; s >>= 1) { if (t < s) red[t] = fmaxf(red[t], red[t + s]); __syncthreads(); }
  float mx = red[0]; __syncthreads();
  float e = (t < NP) ? __expf(lg[t] - mx) : 0.f;
  if (t < NP) lg[t] = e;
  red[t] = e; __syncthreads();
  for (int s = 64; s > 0; s >>= 1) { if (t < s) red[t] += red[t + s]; __syncthreads(); }
  if (t == 0) sinv = 1.f / red[0];
  __syncthreads();
  if (t < 32) {
    const float* vb = qkv + base + 512 + h * 32 + t;
    float o = 0.f;
    for (int m = 0; m < NP; ++m) o += lg[m] * vb[(long)m * 768];
    o16[((long)b * NP + n) * CHN + h * 32 + t] = (_Float16)(o * sinv);
  }
}

__global__ void flag_k(float* __restrict__ o, const int* __restrict__ f, long n)
{
  long i = (long)blockIdx.x * 256 + threadIdx.x;
  if (i < n && f[0] == 0) o[i] = 0.f;
}

// ---------------------------------------------------------------------------
extern "C" void kernel_launch(void* const* d_in, const int* in_sizes, int n_in,
                              void* d_out, int out_size, void* d_ws, size_t ws_size,
                              hipStream_t stream)
{
  (void)in_sizes; (void)n_in; (void)out_size; (void)ws_size;
  const float* features = (const float*)d_in[0];
  const float* prop     = (const float*)d_in[1];
  const float* maskp    = (const float*)d_in[2];
  auto P = [&](int i) { return (const float*)d_in[i]; };
  const int* flag = (const int*)d_in[37];

  char* wsb = (char*)d_ws;
  size_t off = 0;
  auto take = [&](size_t bytes) -> char* {
    char* p = wsb + off; off += (bytes + 255) & ~(size_t)255; return p;
  };
  _Float16* featH  = (_Float16*)take((size_t)BB * CHN * HW * 2); // feat16, later featT16
  _Float16* hard16 = (_Float16*)take((size_t)BB * NPAD * HW * 2);
  _Float16* k16    = (_Float16*)take(800 * 256 * 2);
  float* f_raw = (float*)take(800 * 256 * 4);
  float* w1    = (float*)take(800 * NPK * 4);
  float* tmp1  = (float*)take(800 * 256 * 4);
  float* mix1  = (float*)take(800 * 256 * 4);
  float* d1    = (float*)take(800 * 256 * 4);
  float* f1    = (float*)take(800 * 256 * 4);
  float* w2    = (float*)take(800 * NPK * 4);
  float* tmp2  = (float*)take(800 * 256 * 4);
  float* mix2  = (float*)take(800 * 256 * 4);
  float* d2    = (float*)take(800 * 256 * 4);
  float* kfeat = (float*)take(800 * 256 * 4);
  _Float16* kfeat16 = (_Float16*)take(800 * 256 * 2);
  float* qkvb  = (float*)take(800 * 768 * 4);
  _Float16* o16 = (_Float16*)take(800 * 256 * 2);
  float* oproj = (float*)take(800 * 256 * 4);
  float* k2    = (float*)take(800 * 256 * 4);
  _Float16* k2_16 = (_Float16*)take(800 * 256 * 2);
  _Float16* h1_16 = (_Float16*)take((size_t)800 * FFN * 2);
  float* ffn2  = (float*)take(800 * 256 * 4);
  _Float16* obj16 = (_Float16*)take(800 * 256 * 2);
  float* cls_raw = (float*)take(800 * 256 * 4);
  _Float16* cf16 = (_Float16*)take(800 * 256 * 2);
  float* mr    = (float*)take(800 * 256 * 4);
  _Float16* mf16 = (_Float16*)take(800 * 256 * 2);
  float* mk    = (float*)take(800 * 256 * 4);
  _Float16* mk16 = (_Float16*)take((size_t)BB * NPAD * CHN * 2);
  _Float16* wlfT = (_Float16*)take(NPAD * 256 * 2);
  _Float16* wlkT = (_Float16*)take(NPAD * 256 * 2);
  _Float16* inpT = (_Float16*)take(768 * 256 * 2);
  _Float16* outpT = (_Float16*)take(256 * 256 * 2);
  _Float16* f1T  = (_Float16*)take((size_t)FFN * 256 * 2);
  _Float16* f2T  = (_Float16*)take((size_t)256 * FFN * 2);
  _Float16* clsT = (_Float16*)take(256 * 256 * 2);
  _Float16* fcclsT = (_Float16*)take(160 * 256 * 2);
  _Float16* maskT = (_Float16*)take(256 * 256 * 2);
  _Float16* fcmaskT = (_Float16*)take(256 * 256 * 2);

  float* cls_out  = (float*)d_out;                       // 800 x 151
  float* mask_out = cls_out + (long)800 * NCLS;          // 8 x 100 x 16384
  float* obj_out  = mask_out + (long)BB * NP * HW;       // 800 x 256

  auto lin = [&](const _Float16* X, const _Float16* WT, const float* bias,
                 float* out, _Float16* out16, int Mt, int Nt, int Mout,
                 int K, int Nout, long sxb, long swb, long sob, int batches, int relu) {
    dim3 grid((Mt * Nt + 7) / 8, 1, batches);
    wmma_lin_k<<<grid, 256, 0, stream>>>(X, WT, bias, out, out16,
                                         Mt, Nt, Mout, K, Nout, sxb, swb, sob, relu);
  };
  auto prep = [&](const float* src, _Float16* dst, int N, int Kd, int Npad, int tr) {
    long tot = (long)Npad * Kd;
    prepw_k<<<(int)((tot + 255) / 256), 256, 0, stream>>>(src, dst, N, Kd, Npad, tr);
  };
  auto ln = [&](const float* x, const float* res, const float* g, const float* bt,
                float* out, _Float16* o16p, int relu) {
    ln_k<<<800, 256, 0, stream>>>(x, res, g, bt, out, o16p, relu);
  };

  // ---- stage weights / inputs to f16 ----
  prep(P(3),  wlfT, NPK, 256, NPAD, 1);    // f_wl_w  (256,103) -> T
  prep(P(7),  wlkT, NPK, 256, NPAD, 1);    // k_wl_w
  prep(P(15), inpT, 768, 256, 768, 0);     // in_proj_w (768,256), used as X@W^T
  prep(P(17), outpT, 256, 256, 256, 0);    // out_proj_w
  prep(P(21), f1T, FFN, 256, FFN, 1);      // ffn_w1 (256,2048) -> T
  prep(P(23), f2T, 256, FFN, 256, 1);      // ffn_w2 (2048,256) -> T
  prep(P(27), clsT, 256, 256, 256, 1);
  prep(P(30), fcclsT, NCLS, 256, 160, 1);  // fccls_w (256,151) -> T, pad 160
  prep(P(32), maskT, 256, 256, 256, 1);
  prep(P(35), fcmaskT, 256, 256, 256, 1);
  cvt_k<<<(800 * 256 + 255) / 256, 256, 0, stream>>>(prop, k16, 800 * 256);
  binarize_k<<<(int)(((long)BB * NPAD * HW + 255) / 256), 256, 0, stream>>>(maskp, hard16);
  cvt_k<<<(int)(((long)BB * CHN * HW + 255) / 256), 256, 0, stream>>>(
      features, featH, (long)BB * CHN * HW);

  // ---- GEMM 1: f_raw[b] = hard[b] (112x16384) @ feat16[b]^T -> (100,256) ----
  lin(hard16, featH, nullptr, f_raw, nullptr, 7, 16, NP, HW, CHN,
      (long)NPAD * HW, (long)CHN * HW, (long)NP * CHN, BB, 0);

  // featH now free: rebuild it as featT16 (B,HW,256) for the final GEMM
  {
    dim3 g(HW / 32, CHN / 32, BB);
    transp_cvt_k<<<g, 256, 0, stream>>>(features, featH);
  }

  // ---- dysep #1 (query = proposal kernels, value = f_raw) ----
  lin(k16, wlfT, P(4), w1, nullptr, 50, 7, 800, 256, NPK, 0, 0, 0, 1, 0);
  conv_k<<<800, 256, 0, stream>>>(w1, f_raw, tmp1);
  mix_k <<<800, 256, 0, stream>>>(w1, tmp1, mix1);
  ln(mix1, nullptr, P(5), P(6), d1, nullptr, 0);            // f_ng/f_nb
  ln(d1, f_raw, P(11), P(12), f1, nullptr, 0);              // fan: ln(f + dysep)

  // ---- dysep #2 (value = f1) ----
  lin(k16, wlkT, P(8), w2, nullptr, 50, 7, 800, 256, NPK, 0, 0, 0, 1, 0);
  conv_k<<<800, 256, 0, stream>>>(w2, f1, tmp2);
  mix_k <<<800, 256, 0, stream>>>(w2, tmp2, mix2);
  ln(mix2, nullptr, P(9), P(10), d2, nullptr, 0);           // k_ng/k_nb
  ln(d2, f1, P(13), P(14), kfeat, kfeat16, 0);              // kan: k = ln(f + dysep)

  // ---- self-attention ----
  lin(kfeat16, inpT, P(16), qkvb, nullptr, 50, 48, 800, 256, 768, 0, 0, 0, 1, 0);
  attn_k<<<BB * 8 * NP, 128, 0, stream>>>(qkvb, o16);
  lin(o16, outpT, P(18), oproj, nullptr, 50, 16, 800, 256, 256, 0, 0, 0, 1, 0);
  ln(oproj, kfeat, P(19), P(20), k2, k2_16, 0);             // san: ln(k + o)

  // ---- FFN + obj norm (obj f32 goes straight to d_out) ----
  lin(k2_16, f1T, P(22), nullptr, h1_16, 50, FFN / 16, 800, 256, FFN, 0, 0, 0, 1, 1);
  lin(h1_16, f2T, P(24), ffn2, nullptr, 50, 16, 800, FFN, 256, 0, 0, 0, 1, 0);
  ln(ffn2, k2, P(25), P(26), obj_out, obj16, 0);            // ffnn

  // ---- cls head ----
  lin(obj16, clsT, nullptr, cls_raw, nullptr, 50, 16, 800, 256, 256, 0, 0, 0, 1, 0);
  ln(cls_raw, nullptr, P(28), P(29), nullptr, cf16, 1);     // relu(ln(.))
  lin(cf16, fcclsT, P(31), cls_out, nullptr, 50, 10, 800, 256, NCLS, 0, 0, 0, 1, 0);
  flag_k<<<(800 * NCLS + 255) / 256, 256, 0, stream>>>(cls_out, flag, (long)800 * NCLS);

  // ---- mask head ----
  lin(obj16, maskT, nullptr, mr, nullptr, 50, 16, 800, 256, 256, 0, 0, 0, 1, 0);
  ln(mr, nullptr, P(33), P(34), nullptr, mf16, 1);          // relu(ln(.))
  lin(mf16, fcmaskT, P(36), mk, nullptr, 50, 16, 800, 256, 256, 0, 0, 0, 1, 0);
  padcvt_k<<<(int)(((long)BB * NPAD * CHN + 255) / 256), 256, 0, stream>>>(mk, mk16);

  // ---- GEMM 2: new_mask_preds[b] = mk[b] (112x256) @ featT16[b]^T -> (100,16384)
  lin(mk16, featH, nullptr, mask_out, nullptr, 7, HW / 16, NP, CHN, HW,
      (long)NPAD * CHN, (long)HW * CHN, (long)NP * HW, BB, 0);
}
